// SphereDistanceField_16904991277666
// MI455X (gfx1250) — matrix operand order repaired and checked by reading, not compiled, not run on hardware
//
#include <hip/hip_runtime.h>
#include <hip/hip_bf16.h>
#include <math.h>

typedef __attribute__((ext_vector_type(16))) _Float16 v16h;
typedef __attribute__((ext_vector_type(8)))  float    v8f;

#define N_LEVELS 16
#define TBL (1u << 19)              /* hashmap entries per level */
#define RAYS_PER_BLOCK 128
#define THREADS 256
#define HST 72                      /* activation row stride in halves (144B, 16B aligned) */
#define WST 72                      /* transposed-weight row stride in halves */

__constant__ float c_scale[16] = {
    15.0f,       21.1106061f, 29.5549312f, 41.2242576f,
    57.3502448f, 79.6349552f, 110.430478f, 152.987213f,
    211.796914f, 293.066797f, 405.374691f, 560.574418f,
    775.046928f, 1071.42922f, 1481.00374f, 2047.0f };
__constant__ unsigned c_res[5] = {16u, 23u, 31u, 43u, 59u};   /* dense levels 0..4 */

// ---- WMMA fragment loads from LDS (wave32, 16x16x32 f16) -------------------
// A (16x32, 16-bit): lanes 0-15 hold K{0..7},{16..23}; lanes 16-31 K{8..15},{24..31}
__device__ __forceinline__ v16h load_A(const _Float16* H, int rowBase, int kbase, int lane) {
    const int m = lane & 15, h = lane >> 4;
    const _Float16* p = H + (rowBase + m) * HST + kbase + h * 8;
    v16h a;
    reinterpret_cast<uint4*>(&a)[0] = *reinterpret_cast<const uint4*>(p);
    reinterpret_cast<uint4*>(&a)[1] = *reinterpret_cast<const uint4*>(p + 16);
    return a;
}
// B (32x16, 16-bit): lanes 0-15 hold K{0..7},{16..23} of col N=lane; lanes 16-31 K{8..15},{24..31}
__device__ __forceinline__ v16h load_B(const _Float16* Wt, int ntile, int kbase, int lane) {
    const int m = lane & 15, h = lane >> 4;
    const _Float16* p = Wt + (ntile * 16 + m) * WST + kbase + h * 8;
    v16h b;
    reinterpret_cast<uint4*>(&b)[0] = *reinterpret_cast<const uint4*>(p);
    reinterpret_cast<uint4*>(&b)[1] = *reinterpret_cast<const uint4*>(p + 16);
    return b;
}

__global__ __launch_bounds__(THREADS)
void sdf_fused_kernel(const float* __restrict__ dirs,
                      const float* __restrict__ table,
                      const float* __restrict__ W1, const float* __restrict__ b1,
                      const float* __restrict__ W2, const float* __restrict__ b2,
                      const float* __restrict__ W3, const float* __restrict__ b3,
                      float* __restrict__ out)
{
    __shared__ __align__(16) _Float16 H0[RAYS_PER_BLOCK * HST];
    __shared__ __align__(16) _Float16 H1[RAYS_PER_BLOCK * HST];
    __shared__ __align__(16) _Float16 Wt1[64 * WST];
    __shared__ __align__(16) _Float16 Wt2[64 * WST];
    __shared__ float b1s[64], b2s[64], w3s[64];

    const int t = threadIdx.x;

    // ---- Phase 0: zero activation pad, stage transposed weights into LDS ----
    {
        unsigned long long* z = reinterpret_cast<unsigned long long*>(H0);
        for (int i = t; i < RAYS_PER_BLOCK * HST / 4; i += THREADS) z[i] = 0ull;
        for (int i = t; i < 64 * 64; i += THREADS) {
            const int n = i >> 6, k = i & 63;
            Wt1[n * WST + k] = (k < 35) ? (_Float16)W1[k * 64 + n] : (_Float16)0.f;
            Wt2[n * WST + k] = (_Float16)W2[k * 64 + n];
        }
        if (t < 64) { b1s[t] = b1[t]; b2s[t] = b2[t]; w3s[t] = W3[t]; }
    }
    __syncthreads();

    // ---- Phase 1: multires hash encoding (2 threads per ray, 8 levels each) ----
    {
        const int  rl    = t >> 1;
        const int  which = t & 1;
        const long ray   = (long)blockIdx.x * RAYS_PER_BLOCK + rl;
        const float dx = dirs[ray * 3 + 0], dy = dirs[ray * 3 + 1], dz = dirs[ray * 3 + 2];
        if (which == 0) {
            H0[rl * HST + 0] = (_Float16)dx;
            H0[rl * HST + 1] = (_Float16)dy;
            H0[rl * HST + 2] = (_Float16)dz;
        }
        const float x = fmaf(dx, 0.49f, 0.49f);
        const float y = fmaf(dy, 0.49f, 0.49f);
        const float zc = fmaf(dz, 0.49f, 0.49f);
        #pragma unroll
        for (int i = 0; i < 8; ++i) {
            const int l = which * 8 + i;
            const float s = c_scale[l];
            const float px = fmaf(x, s, 0.5f), py = fmaf(y, s, 0.5f), pz = fmaf(zc, s, 0.5f);
            const float ipx = floorf(px), ipy = floorf(py), ipz = floorf(pz);
            const float fx = px - ipx, fy = py - ipy, fz = pz - ipz;
            const float wx = fx * fx * (3.f - 2.f * fx);
            const float wy = fy * fy * (3.f - 2.f * fy);
            const float wz = fz * fz * (3.f - 2.f * fz);
            const unsigned ix = (unsigned)ipx, iy = (unsigned)ipy, iz = (unsigned)ipz;
            const float vx[2] = {1.f - wx, wx}, vy[2] = {1.f - wy, wy}, vz[2] = {1.f - wz, wz};
            const float* lvl = table + (size_t)l * (size_t)(TBL * 2u);
            float o0 = 0.f, o1 = 0.f;
            #pragma unroll
            for (int c = 0; c < 8; ++c) {
                const unsigned ox = c & 1, oy = (c >> 1) & 1, oz = (c >> 2) & 1;
                const unsigned cx = ix + ox, cy = iy + oy, cz = iz + oz;
                unsigned idx;
                if (l < 5) { const unsigned r = c_res[l]; idx = cx + cy * r + cz * r * r; }
                else       { idx = cx ^ (cy * 2654435761u) ^ (cz * 805459861u); idx &= (TBL - 1u); }
                const float2 f2 = *reinterpret_cast<const float2*>(lvl + (size_t)idx * 2u);
                const float wc = vx[ox] * vy[oy] * vz[oz];
                o0 = fmaf(wc, f2.x, o0);
                o1 = fmaf(wc, f2.y, o1);
            }
            H0[rl * HST + 3 + 2 * l    ] = (_Float16)o0;
            H0[rl * HST + 3 + 2 * l + 1] = (_Float16)o1;
        }
    }
    __syncthreads();

    // ---- Phase 2: MLP via WMMA, one 16-ray M-tile per wave ----
    const int lane    = t & 31;
    const int rowBase = (t >> 5) * 16;
    const int m   = lane & 15;
    const int hlf = lane >> 4;

    // Layer 1: H0[:,0:64] x W1(pad 35->64) -> relu -> H1
    #pragma unroll
    for (int nt = 0; nt < 4; ++nt) {
        v8f acc = {};
        #pragma unroll
        for (int ks = 0; ks < 2; ++ks) {
            v16h a = load_A(H0, rowBase, ks * 32, lane);
            v16h b = load_B(Wt1, nt, ks * 32, lane);
            acc = __builtin_amdgcn_wmma_f32_16x16x32_f16(false, a, false, b,
                                                         (short)0, acc, false, false);
        }
        const int   n    = nt * 16 + m;
        const float bias = b1s[n];
        #pragma unroll
        for (int i = 0; i < 8; ++i) {
            float v = acc[i] + bias;
            v = v > 0.f ? v : 0.f;
            H1[(rowBase + i + hlf * 8) * HST + n] = (_Float16)v;
        }
    }
    __syncthreads();

    // Layer 2: H1 x W2 -> relu -> H0 (reused)
    #pragma unroll
    for (int nt = 0; nt < 4; ++nt) {
        v8f acc = {};
        #pragma unroll
        for (int ks = 0; ks < 2; ++ks) {
            v16h a = load_A(H1, rowBase, ks * 32, lane);
            v16h b = load_B(Wt2, nt, ks * 32, lane);
            acc = __builtin_amdgcn_wmma_f32_16x16x32_f16(false, a, false, b,
                                                         (short)0, acc, false, false);
        }
        const int   n    = nt * 16 + m;
        const float bias = b2s[n];
        #pragma unroll
        for (int i = 0; i < 8; ++i) {
            float v = acc[i] + bias;
            v = v > 0.f ? v : 0.f;
            H0[(rowBase + i + hlf * 8) * HST + n] = (_Float16)v;
        }
    }
    __syncthreads();

    // ---- Phase 3: layer 3 (64 -> 1) + softplus, one thread per ray ----
    if (t < RAYS_PER_BLOCK) {
        float sum = 0.f;
        const _Float16* hr = H0 + t * HST;
        #pragma unroll
        for (int c = 0; c < 8; ++c) {
            uint4 pk = *reinterpret_cast<const uint4*>(hr + c * 8);
            const _Float16* hp = reinterpret_cast<const _Float16*>(&pk);
            #pragma unroll
            for (int j = 0; j < 8; ++j)
                sum = fmaf((float)hp[j], w3s[c * 8 + j], sum);
        }
        const float xo   = sum + b3[0] + 1.0f;
        const float dist = (xo > 20.f) ? xo : log1pf(expf(xo));
        out[(long)blockIdx.x * RAYS_PER_BLOCK + t] = dist;
    }
}

extern "C" void kernel_launch(void* const* d_in, const int* in_sizes, int n_in,
                              void* d_out, int out_size, void* d_ws, size_t ws_size,
                              hipStream_t stream) {
    const float* dirs  = (const float*)d_in[0];
    const float* table = (const float*)d_in[1];
    const float* W1    = (const float*)d_in[2];
    const float* b1    = (const float*)d_in[3];
    const float* W2    = (const float*)d_in[4];
    const float* b2    = (const float*)d_in[5];
    const float* W3    = (const float*)d_in[6];
    const float* b3    = (const float*)d_in[7];
    float* out = (float*)d_out;

    const int nrays = out_size;   // 2,097,152
    const int grid  = (nrays + RAYS_PER_BLOCK - 1) / RAYS_PER_BLOCK;
    sdf_fused_kernel<<<grid, THREADS, 0, stream>>>(dirs, table, W1, b1, W2, b2, W3, b3, out);
}
